// Model_46042049413586
// MI455X (gfx1250) — matrix operand order repaired and checked by reading, not compiled, not run on hardware
//
#include <hip/hip_runtime.h>

#define DEV_INLINE __device__ __forceinline__

typedef __bf16 v16bf __attribute__((ext_vector_type(16)));
typedef float v8f __attribute__((ext_vector_type(8)));
typedef unsigned int u32x4 __attribute__((ext_vector_type(4)));
typedef unsigned short us4 __attribute__((ext_vector_type(4)));
typedef unsigned short ushort_t;

constexpr int B_ = 32;
constexpr int T_ = 1024;
constexpr int D_ = 512;
constexpr int M_ = B_ * T_;          // 32768 rows of the (B*T, D) activation matrix
constexpr int CIN = 21;
constexpr int MARK_ = 4;
constexpr int PRED = 96;
constexpr float LN1E4 = 9.2103403719761836f;   // ln(10000)

DEV_INLINE ushort_t f2bf(float f) {            // fp32 -> bf16 round-to-nearest-even
  union { float f; unsigned u; } v; v.f = f;
  unsigned r = v.u + 0x7FFFu + ((v.u >> 16) & 1u);
  return (ushort_t)(r >> 16);
}
DEV_INLINE float bf2f(ushort_t h) {
  union { unsigned u; float f; } v; v.u = (unsigned)h << 16; return v.f;
}
DEV_INLINE float sigmoidf(float x) { return 1.0f / (1.0f + __expf(-x)); }

// ---------------------------------------------------------------------------
// Per (batch, channel) mean / std over the time axis.
// ---------------------------------------------------------------------------
__global__ __launch_bounds__(128)
void stats_kernel(const float* __restrict__ x_enc, float* __restrict__ mean,
                  float* __restrict__ stdv) {
  int bc = blockIdx.x;
  int b = bc / CIN, c = bc % CIN;
  const float* p = x_enc + (size_t)b * T_ * CIN + c;
  float s = 0.f, s2 = 0.f;
  for (int t = threadIdx.x; t < T_; t += 128) {
    float v = p[(size_t)t * CIN];
    s += v; s2 += v * v;
  }
  __shared__ float rs[128], rq[128];
  rs[threadIdx.x] = s; rq[threadIdx.x] = s2;
  __syncthreads();
  for (int o = 64; o > 0; o >>= 1) {
    if (threadIdx.x < o) { rs[threadIdx.x] += rs[threadIdx.x + o]; rq[threadIdx.x] += rq[threadIdx.x + o]; }
    __syncthreads();
  }
  if (threadIdx.x == 0) {
    float mu  = rs[0] / T_;
    float var = rq[0] / T_ - mu * mu;
    mean[bc] = mu;
    stdv[bc] = sqrtf(var + 1e-5f);
  }
}

// ---------------------------------------------------------------------------
// Fused: normalize + wrap-pad conv1d(k=3) + positional emb + temporal emb.
// ---------------------------------------------------------------------------
__global__ __launch_bounds__(128)
void embed_kernel(const float* __restrict__ x_enc, const float* __restrict__ x_mark,
                  const float* __restrict__ conv_w, const float* __restrict__ temp_w,
                  const float* __restrict__ mean, const float* __restrict__ stdv,
                  float* __restrict__ x) {
  int row = blockIdx.x;                // b*T + t
  int b = row >> 10, t = row & 1023;
  __shared__ float xc[3 * CIN];
  __shared__ float mk[MARK_];
  int tid = threadIdx.x;
  if (tid < 3 * CIN) {
    int w = tid / CIN, c = tid % CIN;
    int tt = (t + w - 1 + T_) & (T_ - 1);            // wrap padding
    float v = x_enc[((size_t)b * T_ + tt) * CIN + c];
    xc[tid] = (v - mean[b * CIN + c]) / stdv[b * CIN + c];
  } else if (tid < 3 * CIN + MARK_) {
    int m = tid - 3 * CIN;
    mk[m] = x_mark[((size_t)b * T_ + t) * MARK_ + m];
  }
  __syncthreads();
  float* xrow = x + (size_t)row * D_;
#pragma unroll
  for (int r = 0; r < 4; ++r) {
    int d = tid + r * 128;
    float acc = 0.f;
    for (int i = 0; i < 3 * CIN; ++i) acc += xc[i] * conv_w[(size_t)i * D_ + d];
#pragma unroll
    for (int m = 0; m < MARK_; ++m)   acc += mk[m] * temp_w[(size_t)m * D_ + d];
    int i2 = d >> 1;
    float freq = __expf(-(float)i2 * (LN1E4 / 256.f));
    float sv, cv; __sincosf((float)t * freq, &sv, &cv);
    acc += (d & 1) ? cv : sv;
    xrow[d] = acc;
  }
}

// ---------------------------------------------------------------------------
// LayerNorm over D=512: fp32 in, bf16 out (GEMM A operand).
// ---------------------------------------------------------------------------
__global__ __launch_bounds__(128)
void layernorm_kernel(const float* __restrict__ x, const float* __restrict__ w,
                      const float* __restrict__ bb, ushort_t* __restrict__ xn) {
  int row = blockIdx.x;
  float4 v = ((const float4*)(x + (size_t)row * D_))[threadIdx.x];
  __shared__ float red[128];
  red[threadIdx.x] = v.x + v.y + v.z + v.w;
  __syncthreads();
  for (int o = 64; o > 0; o >>= 1) { if (threadIdx.x < o) red[threadIdx.x] += red[threadIdx.x + o]; __syncthreads(); }
  float mu = red[0] * (1.f / D_);
  __syncthreads();
  float dx = v.x - mu, dy = v.y - mu, dz = v.z - mu, dw = v.w - mu;
  red[threadIdx.x] = dx * dx + dy * dy + dz * dz + dw * dw;
  __syncthreads();
  for (int o = 64; o > 0; o >>= 1) { if (threadIdx.x < o) red[threadIdx.x] += red[threadIdx.x + o]; __syncthreads(); }
  float rstd = rsqrtf(red[0] * (1.f / D_) + 1e-5f);
  int c0 = threadIdx.x * 4;
  us4 o4;
  o4[0] = f2bf(dx * rstd * w[c0 + 0] + bb[c0 + 0]);
  o4[1] = f2bf(dy * rstd * w[c0 + 1] + bb[c0 + 1]);
  o4[2] = f2bf(dz * rstd * w[c0 + 2] + bb[c0 + 2]);
  o4[3] = f2bf(dw * rstd * w[c0 + 3] + bb[c0 + 3]);
  ((us4*)(xn + (size_t)row * D_))[threadIdx.x] = o4;
}

// ---------------------------------------------------------------------------
// Weight convert+transpose: W[K=512][N=512] fp32 -> Wt[N][K] bf16.
// 32x32 LDS tile, coalesced both directions.
// ---------------------------------------------------------------------------
__global__ __launch_bounds__(256)
void wt_bf16_kernel(const float* __restrict__ W, ushort_t* __restrict__ Wt) {
  __shared__ ushort_t tile[32][33];
  int k0 = blockIdx.x * 32, n0 = blockIdx.y * 32;
  int tx = threadIdx.x & 31;
  int ty0 = threadIdx.x >> 5;          // 0..7
  for (int ty = ty0; ty < 32; ty += 8)
    tile[tx][ty] = f2bf(W[(size_t)(k0 + ty) * D_ + n0 + tx]);   // tile[n][k]
  __syncthreads();
  for (int ty = ty0; ty < 32; ty += 8)
    Wt[(size_t)(n0 + ty) * D_ + k0 + tx] = tile[ty][tx];
}

// ---------------------------------------------------------------------------
// bf16 WMMA GEMM: C = A[M,512](bf16) * Wt[N,K](bf16, pre-transposed)
//   (+bias fp32) (+residual fp32) -> fp32 or bf16 output.
// Block = 256 threads (8 wave32), block tile 128x64, wave tile 32x32 (2x2 WMMA).
// Inner k-step = 8 x global_load_b128 + 4 x v_wmma_f32_16x16x32_bf16.
// Fragment layouts per CDNA5 ISA:
//   A 16x32: lane-half h -> chunks K=[8h..8h+7] (VGPR0-3), [16+8h..] (VGPR4-7)
//   B 32x16: lane = N col, chunks K=[8h..8h+7] (VGPR0-3), [16+8h..] (VGPR4-7)
// ---------------------------------------------------------------------------
union FragCvt { u32x4 q[2]; v16bf v; };

template<bool HAS_BIAS, bool HAS_RES, bool OUT_F32>
__global__ __launch_bounds__(256)
void gemm_wmma(const ushort_t* __restrict__ A, const ushort_t* __restrict__ Wt,
               const float* __restrict__ bias, const float* __restrict__ residual,
               void* __restrict__ Cv) {
  constexpr int K = 512, N = 512;
  const int lane = threadIdx.x & 31;
  const int wave = threadIdx.x >> 5;
  const int m0 = blockIdx.y * 128 + (wave >> 1) * 32;
  const int n0 = blockIdx.x * 64 + (wave & 1) * 32;
  const int hl = lane >> 4;
  const int lr = lane & 15;

  const ushort_t* arow0 = A  + (size_t)(m0 + lr) * K;
  const ushort_t* arow1 = arow0 + (size_t)16 * K;
  const ushort_t* bcol0 = Wt + (size_t)(n0 + lr) * K;
  const ushort_t* bcol1 = bcol0 + (size_t)16 * K;
  const int aoff = 8 * hl;
  const int boff = 8 * hl;

  v8f acc00 = {}, acc01 = {}, acc10 = {}, acc11 = {};

#pragma unroll 2
  for (int k0 = 0; k0 < K; k0 += 32) {
    if (k0 + 32 < K) {                               // -> global_prefetch_b8
      __builtin_prefetch(arow0 + k0 + 32, 0, 1);
      __builtin_prefetch(bcol0 + k0 + 32, 0, 1);
    }
    FragCvt a0, a1, b0, b1;
    a0.q[0] = *(const u32x4*)(arow0 + k0 + aoff);
    a0.q[1] = *(const u32x4*)(arow0 + k0 + 16 + aoff);
    a1.q[0] = *(const u32x4*)(arow1 + k0 + aoff);
    a1.q[1] = *(const u32x4*)(arow1 + k0 + 16 + aoff);
    b0.q[0] = *(const u32x4*)(bcol0 + k0 + boff);
    b0.q[1] = *(const u32x4*)(bcol0 + k0 + 16 + boff);
    b1.q[0] = *(const u32x4*)(bcol1 + k0 + boff);
    b1.q[1] = *(const u32x4*)(bcol1 + k0 + 16 + boff);
    acc00 = __builtin_amdgcn_wmma_f32_16x16x32_bf16(false, a0.v, false, b0.v, (short)0, acc00, false, false);
    acc01 = __builtin_amdgcn_wmma_f32_16x16x32_bf16(false, a0.v, false, b1.v, (short)0, acc01, false, false);
    acc10 = __builtin_amdgcn_wmma_f32_16x16x32_bf16(false, a1.v, false, b0.v, (short)0, acc10, false, false);
    acc11 = __builtin_amdgcn_wmma_f32_16x16x32_bf16(false, a1.v, false, b1.v, (short)0, acc11, false, false);
  }

  float* Cf = (float*)Cv;
  ushort_t* Ch = (ushort_t*)Cv;
  auto store_tile = [&](v8f acc, int mo, int no) {
#pragma unroll
    for (int i = 0; i < 8; ++i) {
      int row = m0 + mo + hl * 8 + i;       // C/D layout: VGPR i -> M = 8*half + i
      int col = n0 + no + lr;               // lane -> N
      float val = acc[i];
      if constexpr (HAS_BIAS) val += bias[col];
      size_t idx = (size_t)row * N + col;
      if constexpr (HAS_RES) val += residual[idx];
      if constexpr (OUT_F32) Cf[idx] = val;
      else                   Ch[idx] = f2bf(val);
    }
  };
  store_tile(acc00, 0, 0);
  store_tile(acc01, 0, 16);
  store_tile(acc10, 16, 0);
  store_tile(acc11, 16, 16);
}

// ---------------------------------------------------------------------------
// Fused RoPE + sigmoid gate + q.k dot -> scores[b,t] (q/k never materialized).
// ---------------------------------------------------------------------------
__global__ __launch_bounds__(256)
void scores_kernel(const ushort_t* __restrict__ Zq, const ushort_t* __restrict__ Zk,
                   const ushort_t* __restrict__ Gq, const ushort_t* __restrict__ Gk,
                   float* __restrict__ scores) {
  int row = blockIdx.x;
  int t = row & 1023;
  int i = threadIdx.x;                       // half-dim index 0..255
  const ushort_t* zq = Zq + (size_t)row * D_;
  const ushort_t* zk = Zk + (size_t)row * D_;
  const ushort_t* gq = Gq + (size_t)row * D_;
  const ushort_t* gk = Gk + (size_t)row * D_;
  float freq = __expf(-(float)i * (LN1E4 / 256.f));
  float sv, cv; __sincosf((float)t * freq, &sv, &cv);
  float q1 = bf2f(zq[i]), q2 = bf2f(zq[i + 256]);
  float k1 = bf2f(zk[i]), k2 = bf2f(zk[i + 256]);
  float qa = (q1 * cv - q2 * sv) * sigmoidf(bf2f(gq[i]));
  float qb = (q1 * sv + q2 * cv) * sigmoidf(bf2f(gq[i + 256]));
  float ka = (k1 * cv - k2 * sv) * sigmoidf(bf2f(gk[i]));
  float kb = (k1 * sv + k2 * cv) * sigmoidf(bf2f(gk[i + 256]));
  __shared__ float red[256];
  red[i] = qa * ka + qb * kb;
  __syncthreads();
  for (int o = 128; o > 0; o >>= 1) { if (i < o) red[i] += red[i + o]; __syncthreads(); }
  if (i == 0) scores[row] = red[0] * 0.044194173824159216f;   // 1/sqrt(512)
}

// Softmax over the BATCH axis (axis=0 in the reference). One thread per t.
__global__ __launch_bounds__(256)
void softmax_b_kernel(const float* __restrict__ scores, float* __restrict__ a) {
  int t = blockIdx.x * 256 + threadIdx.x;
  float mx = -1e30f;
  for (int b = 0; b < B_; ++b) mx = fmaxf(mx, scores[(size_t)b * T_ + t]);
  float sum = 0.f;
  for (int b = 0; b < B_; ++b) sum += __expf(scores[(size_t)b * T_ + t] - mx);
  float inv = 1.0f / sum;
  for (int b = 0; b < B_; ++b)
    a[(size_t)b * T_ + t] = __expf(scores[(size_t)b * T_ + t] - mx) * inv;
}

// Linear recurrence h = a_t*h + B_t (fp32 state); out = h*v_t (bf16 I/O).
// One block per batch, 512 threads (one per d); a[b,:] staged in LDS.
__global__ __launch_bounds__(512)
void scan_kernel(const float* __restrict__ a, const ushort_t* __restrict__ Bx,
                 const ushort_t* __restrict__ V, ushort_t* __restrict__ out) {
  int b = blockIdx.x;
  int d = threadIdx.x;
  __shared__ float ash[T_];
  for (int t = threadIdx.x; t < T_; t += 512) ash[t] = a[(size_t)b * T_ + t];
  __syncthreads();
  const ushort_t* bx = Bx + (size_t)b * T_ * D_ + d;
  const ushort_t* vv = V  + (size_t)b * T_ * D_ + d;
  ushort_t* op = out + (size_t)b * T_ * D_ + d;
  float h = 0.f;
  for (int t = 0; t < T_; ++t) {
    h = ash[t] * h + bf2f(bx[(size_t)t * D_]);
    op[(size_t)t * D_] = f2bf(h * bf2f(vv[(size_t)t * D_]));
  }
}

// Final projection 512->21, de-normalize, keep last PRED steps.
__global__ __launch_bounds__(256)
void proj_kernel(const float* __restrict__ x, const float* __restrict__ pw,
                 const float* __restrict__ pb, const float* __restrict__ mean,
                 const float* __restrict__ stdv, float* __restrict__ out) {
  int idx = blockIdx.x * 256 + threadIdx.x;
  if (idx >= B_ * PRED * CIN) return;
  int c = idx % CIN;
  int r = idx / CIN;
  int tt = (r % PRED) + (T_ - PRED);
  int b = r / PRED;
  const float* xr = x + ((size_t)b * T_ + tt) * D_;
  float acc = pb[c];
  for (int dd = 0; dd < D_; ++dd) acc += xr[dd] * pw[(size_t)dd * CIN + c];
  out[idx] = acc * stdv[b * CIN + c] + mean[b * CIN + c];
}

// ---------------------------------------------------------------------------
extern "C" void kernel_launch(void* const* d_in, const int* in_sizes, int n_in,
                              void* d_out, int out_size, void* d_ws, size_t ws_size,
                              hipStream_t stream) {
  (void)in_sizes; (void)n_in; (void)out_size; (void)ws_size;
  const float* x_enc  = (const float*)d_in[0];
  const float* x_mark = (const float*)d_in[1];
  const float* conv_w = (const float*)d_in[4];
  const float* temp_w = (const float*)d_in[5];
  const float* ln_w   = (const float*)d_in[6];
  const float* ln_b   = (const float*)d_in[7];
  const float* gate_b = (const float*)d_in[12];
  const float* Bb     = (const float*)d_in[14];
  const float* ob     = (const float*)d_in[16];
  const float* proj_w = (const float*)d_in[17];
  const float* proj_b = (const float*)d_in[18];
  const float* Wsrc[6] = { (const float*)d_in[8],  (const float*)d_in[9],
                           (const float*)d_in[10], (const float*)d_in[11],
                           (const float*)d_in[13], (const float*)d_in[15] };
  float* out = (float*)d_out;

  char* wsb = (char*)d_ws;
  size_t off = 0;
  auto allocB = [&](size_t bytes) { void* p = wsb + off; off += (bytes + 255) & ~(size_t)255; return p; };
  constexpr size_t MD = (size_t)M_ * D_;
  float*    meanp   = (float*)allocB(B_ * CIN * 4);
  float*    stdp    = (float*)allocB(B_ * CIN * 4);
  float*    scoresp = (float*)allocB((size_t)B_ * T_ * 4);
  float*    ap      = (float*)allocB((size_t)B_ * T_ * 4);
  float*    x       = (float*)allocB(MD * 4);
  ushort_t* xn      = (ushort_t*)allocB(MD * 2);
  ushort_t* Zq      = (ushort_t*)allocB(MD * 2);
  ushort_t* Zk      = (ushort_t*)allocB(MD * 2);
  ushort_t* Vv      = (ushort_t*)allocB(MD * 2);
  ushort_t* Gq      = (ushort_t*)allocB(MD * 2);
  ushort_t* Gk      = (ushort_t*)allocB(MD * 2);
  ushort_t* Bx      = (ushort_t*)allocB(MD * 2);
  ushort_t* WtBase  = (ushort_t*)allocB((size_t)18 * D_ * D_ * 2);

  // Pre-convert + transpose all 18 weight matrices to bf16 Wt[N][K].
  dim3 tgrid(D_ / 32, D_ / 32);
  for (int l = 0; l < 3; ++l)
    for (int s = 0; s < 6; ++s)
      wt_bf16_kernel<<<tgrid, 256, 0, stream>>>(Wsrc[s] + (size_t)l * D_ * D_,
                                                WtBase + (size_t)(l * 6 + s) * D_ * D_);

  stats_kernel<<<B_ * CIN, 128, 0, stream>>>(x_enc, meanp, stdp);
  embed_kernel<<<M_, 128, 0, stream>>>(x_enc, x_mark, conv_w, temp_w, meanp, stdp, x);

  dim3 ggrid(D_ / 64, M_ / 128);     // (8, 256) block tiles of 128x64
  for (int l = 0; l < 3; ++l) {
    size_t bOff = (size_t)l * D_;
    const ushort_t* WtQ = WtBase + (size_t)(l * 6 + 0) * D_ * D_;
    const ushort_t* WtK = WtBase + (size_t)(l * 6 + 1) * D_ * D_;
    const ushort_t* WtV = WtBase + (size_t)(l * 6 + 2) * D_ * D_;
    const ushort_t* WtG = WtBase + (size_t)(l * 6 + 3) * D_ * D_;
    const ushort_t* WtB = WtBase + (size_t)(l * 6 + 4) * D_ * D_;
    const ushort_t* WtO = WtBase + (size_t)(l * 6 + 5) * D_ * D_;

    layernorm_kernel<<<M_, 128, 0, stream>>>(x, ln_w + bOff, ln_b + bOff, xn);
    gemm_wmma<false, false, false><<<ggrid, 256, 0, stream>>>(xn, WtQ, nullptr, nullptr, Zq);
    gemm_wmma<false, false, false><<<ggrid, 256, 0, stream>>>(xn, WtK, nullptr, nullptr, Zk);
    gemm_wmma<false, false, false><<<ggrid, 256, 0, stream>>>(xn, WtV, nullptr, nullptr, Vv);
    gemm_wmma<true,  false, false><<<ggrid, 256, 0, stream>>>(Zq, WtG, gate_b + bOff, nullptr, Gq);
    gemm_wmma<true,  false, false><<<ggrid, 256, 0, stream>>>(Zk, WtG, gate_b + bOff, nullptr, Gk);
    gemm_wmma<true,  false, false><<<ggrid, 256, 0, stream>>>(xn, WtB, Bb + bOff, nullptr, Bx);
    scores_kernel<<<M_, 256, 0, stream>>>(Zq, Zk, Gq, Gk, scoresp);
    softmax_b_kernel<<<T_ / 256, 256, 0, stream>>>(scoresp, ap);
    scan_kernel<<<B_, 512, 0, stream>>>(ap, Bx, Vv, Zq);           // Zq reused as scan output
    gemm_wmma<true, true, true><<<ggrid, 256, 0, stream>>>(Zq, WtO, ob + bOff, x, x);  // +bias +residual -> fp32 x
  }
  proj_kernel<<<(B_ * PRED * CIN + 255) / 256, 256, 0, stream>>>(x, proj_w, proj_b, meanp, stdp, out);
}
